// InnerProductFeatures_88252987998529
// MI455X (gfx1250) — matrix operand order repaired and checked by reading, not compiled, not run on hardware
//
#include <hip/hip_runtime.h>
#include <hip/hip_bf16.h>
#include <cmath>

typedef __attribute__((ext_vector_type(16))) __bf16 v16bf;
typedef __attribute__((ext_vector_type(8)))  float  v8f;
typedef __attribute__((ext_vector_type(4)))  float  v4f;

#define NB   65536   // batches
#define NC   64      // channels
#define ND   32      // features

// ---------------------------------------------------------------------------
// Prep: swizzle W [64][32][32] f32 into per-lane bf16 B-fragments in ws.
// Fragment (j, h) feeds v_wmma_f32_16x16x32_bf16 with B[K=m][N=n] = W[j, h*16+n, m].
// 16-bit B 32x16 layout: lane<16 holds column N=lane, K=0..7 and 16..23 (2/VGPR);
// lane>=16 holds N=lane-16, K=8..15 and 24..31.
// ws element index = ((j*2 + h)*32 + lane)*16 + t,  value = bf16(W[j][h*16+(lane&15)][((lane>>4)<<3)+(t&7)+((t>>3)<<4)])
// ---------------------------------------------------------------------------
__global__ void prep_W_kernel(const float* __restrict__ W, __bf16* __restrict__ wsW) {
    int idx  = blockIdx.x * blockDim.x + threadIdx.x;   // 0..65535
    int t    = idx & 15;
    int lane = (idx >> 4) & 31;
    int h    = (idx >> 9) & 1;
    int j    = idx >> 10;
    int k = h * 16 + (lane & 15);
    int m = ((lane >> 4) << 3) + (t & 7) + ((t >> 3) << 4);
    wsW[idx] = (__bf16)W[(j * ND + k) * ND + m];
}

// ---------------------------------------------------------------------------
// Main: one wave per 16-batch tile.
// Step 1: S[16x32] = sum_j  x_tile[16x32]_j  @  W[j]^T   via bf16 WMMA (f32 acc)
// Step 2: out[b,i] = tanh( dot(x[b,i,:], S[b,:]) )        via VALU (L2 re-read)
// ---------------------------------------------------------------------------
__global__ __launch_bounds__(256) void ipf_kernel(const float* __restrict__ x,
                                                  const __bf16* __restrict__ wsW,
                                                  float* __restrict__ out) {
    __shared__ float s_tile[8][16][32];   // per-wave S tiles (16 KB)

    const int lane = threadIdx.x & 31;
    const int wave = threadIdx.x >> 5;
    const int tile = blockIdx.x * 8 + wave;          // 0..4095
    const size_t b0 = (size_t)tile * 16;

    // A-fragment per-lane addressing (16-bit A 16x32 layout)
    const int row  = lane & 15;                      // M = batch within tile
    const int hsel = lane >> 4;
    const int kb   = hsel * 8;                       // K runs: [kb,kb+8) and [kb+16,kb+24)

    v8f acc0 = {};                                   // S[:, 0..15]
    v8f acc1 = {};                                   // S[:,16..31]

    const float* xrow = x + (b0 + row) * (NC * ND);
    const v16bf* wb   = (const v16bf*)wsW;

    for (int j = 0; j < NC; ++j) {
        const float* xp = xrow + j * ND;
        v4f r0 = *(const v4f*)(xp + kb);
        v4f r1 = *(const v4f*)(xp + kb + 4);
        v4f r2 = *(const v4f*)(xp + kb + 16);
        v4f r3 = *(const v4f*)(xp + kb + 20);
        v16bf a;
#pragma unroll
        for (int e = 0; e < 4; ++e) {
            a[e]      = (__bf16)r0[e];
            a[4 + e]  = (__bf16)r1[e];
            a[8 + e]  = (__bf16)r2[e];
            a[12 + e] = (__bf16)r3[e];
        }
        v16bf bf0 = wb[(j * 2 + 0) * 32 + lane];
        v16bf bf1 = wb[(j * 2 + 1) * 32 + lane];
        acc0 = __builtin_amdgcn_wmma_f32_16x16x32_bf16(false, a, false, bf0,
                                                       (short)0, acc0, false, false);
        acc1 = __builtin_amdgcn_wmma_f32_16x16x32_bf16(false, a, false, bf1,
                                                       (short)0, acc1, false, false);
    }

    // Scatter S to LDS per C/D layout: VGPR r, lane l -> M=(l<16?r:r+8), N=l&15
    {
        const int kc = lane & 15;
        const int bb = hsel * 8;
#pragma unroll
        for (int r = 0; r < 8; ++r) {
            s_tile[wave][bb + r][kc]      = acc0[r];
            s_tile[wave][bb + r][kc + 16] = acc1[r];
        }
    }
    __syncthreads();

    // Step 2: lane -> (batch = lane&15, channel half = lane>>4); 32 channels each.
    {
        const int bl  = lane & 15;
        const int ch0 = hsel * 32;
        float s[ND];
#pragma unroll
        for (int k = 0; k < ND; ++k) s[k] = s_tile[wave][bl][k];   // broadcast reads

        const float* xb = x   + (b0 + bl) * (NC * ND);
        float*       ob = out + (b0 + bl) * NC;
        for (int ic = 0; ic < 32; ++ic) {
            const int i = ch0 + ic;
            const float* xp = xb + i * ND;
            float dot = 0.f;
#pragma unroll
            for (int m = 0; m < ND; m += 4) {
                v4f xv = *(const v4f*)(xp + m);
                dot += xv[0] * s[m] + xv[1] * s[m + 1] + xv[2] * s[m + 2] + xv[3] * s[m + 3];
            }
            ob[i] = tanhf(dot);
        }
    }
}

extern "C" void kernel_launch(void* const* d_in, const int* in_sizes, int n_in,
                              void* d_out, int out_size, void* d_ws, size_t ws_size,
                              hipStream_t stream) {
    const float* x = (const float*)d_in[0];   // [65536, 64, 32] f32
    const float* W = (const float*)d_in[1];   // [64, 32, 32]    f32
    float* out     = (float*)d_out;           // [65536, 64]     f32
    __bf16* wsW    = (__bf16*)d_ws;           // 128 KB B-fragment cache

    prep_W_kernel<<<256, 256, 0, stream>>>(W, wsW);          // 65536 threads
    ipf_kernel<<<NB / 16 / 8, 256, 0, stream>>>(x, wsW, out); // 512 blocks x 8 waves
}
